// RNN_60799557042628
// MI455X (gfx1250) — matrix operand order repaired and checked by reading, not compiled, not run on hardware
//
#include <hip/hip_runtime.h>
#include <hip/hip_bf16.h>
#include <math.h>
#include <stdint.h>

typedef __bf16 bf16_t;
typedef __attribute__((ext_vector_type(16))) __bf16 v16bf;
typedef __attribute__((ext_vector_type(8)))  __bf16 v8bf;
typedef __attribute__((ext_vector_type(8)))  float  v8f;

#define N_EMBD 512
#define N_HID  1024
#define VOCAB  32000
#define BATCH  8
#define SEQ    512
#define BT     (BATCH * SEQ)   // 4096
#define G3     (3 * N_HID)     // 3072

// ---------------------------------------------------------------------------
// Pack a row-major f32 matrix W[K][N] (ld = row stride) into the CDNA5 WMMA
// bf16 B-fragment layout:  dst[((nt*nkc + kc)*32 + lane)*16 + i] =
//   W[kc*32 + (i/8)*16 + (lane/16)*8 + i%8][nt*16 + lane%16]
// One 16x16 N-tile / 32-K chunk = 1KB contiguous, 32B per lane.
// ---------------------------------------------------------------------------
__global__ void pack_b_kernel(const float* __restrict__ src, int ld, int K, int N,
                              bf16_t* __restrict__ dst) {
    size_t total = (size_t)K * N;
    size_t o = (size_t)blockIdx.x * blockDim.x + threadIdx.x;
    if (o >= total) return;
    int i    = (int)(o & 15);
    int lane = (int)((o >> 4) & 31);
    int nkc  = K >> 5;
    int kc   = (int)((o >> 9) % nkc);
    int nt   = (int)(o / ((size_t)nkc << 9));
    int k = kc * 32 + ((i >> 3) << 4) + ((lane >> 4) << 3) + (i & 7);
    int n = nt * 16 + (lane & 15);
    dst[o] = (bf16_t)src[(size_t)k * ld + n];
}

// Embedding gather + cast to bf16: embB[bt][e] = wte[idx[bt]][e]
__global__ void embed_kernel(const int* __restrict__ idx, const float* __restrict__ wte,
                             bf16_t* __restrict__ embB) {
    size_t o = (size_t)blockIdx.x * blockDim.x + threadIdx.x;
    if (o >= (size_t)BT * N_EMBD) return;
    int bt = (int)(o >> 9);
    int e  = (int)(o & (N_EMBD - 1));
    int tok = idx[bt];
    embB[o] = (bf16_t)wte[(size_t)tok * N_EMBD + e];
}

// A-fragment from row-major bf16 (16-bit A 16x32 layout): two 16B loads/lane.
__device__ __forceinline__ v16bf load_a_frag(const bf16_t* __restrict__ arow, int kc) {
    v8bf lo = *(const v8bf*)(arow + kc * 32);
    v8bf hi = *(const v8bf*)(arow + kc * 32 + 16);
    return __builtin_shufflevector(lo, hi, 0,1,2,3,4,5,6,7,8,9,10,11,12,13,14,15);
}

// ---------------------------------------------------------------------------
// Staged bf16 WMMA GEMM: C[M][N] = A[M][K] @ Bpacked + bias.
//  - block = 256 threads = 8 waves; block covers 8 M-tiles x 4 N-tiles
//  - B chunks (4 tiles x 1KB per K=32 slice) are DMA'd into LDS with
//    global_load_async_to_lds_b128 (ASYNCcnt), double-buffered:
//       issue chunk kc+1 -> wait chunk kc (s_wait_asynccnt) -> barrier ->
//       8 waves compute 32 WMMAs from the 4KB LDS buffer -> barrier
//  - each wave holds 4 f32 accumulators and amortizes one A fragment over
//    4 v_wmma_f32_16x16x32_bf16 per chunk.
// Requires: Mtiles % 8 == 0, Ntiles % 4 == 0 (exact grids, no early-outs).
// ---------------------------------------------------------------------------
__global__ void __launch_bounds__(256, 2)
gemm_wmma_staged_kernel(const bf16_t* __restrict__ A, int lda,
                        const bf16_t* __restrict__ Bp,
                        const float* __restrict__ bias,
                        float* __restrict__ C, int ldc,
                        int Mtiles, int Ngroups, int nkc) {
    __shared__ char ldsB[2 * 4 * 1024];   // double buffer, 4 tiles x 1KB

    int lane = threadIdx.x & 31;
    int wave = threadIdx.x >> 5;          // 0..7
    int mgroups = Mtiles >> 3;
    int mtBase  = (blockIdx.x % mgroups) << 3;
    int ng      = blockIdx.x / mgroups;   // < Ngroups
    int mt      = mtBase + wave;
    (void)Ngroups;

    const bf16_t* arow = A + (size_t)(mt * 16 + (lane & 15)) * lda + ((lane >> 4) << 3);
    const char*   bbase = (const char*)Bp;   // byte view of packed B
    unsigned ldsBase = (unsigned)(uintptr_t)&ldsB[0];

    // Prologue: wave 0 DMAs chunk 0 into buffer 0 (8 async b128 issues).
    if (wave == 0) {
#pragma unroll
        for (int j = 0; j < 4; ++j) {
            uint64_t g0 = (uint64_t)(uintptr_t)(bbase +
                          (((size_t)(ng * 4 + j) * nkc) << 10)) + (unsigned)(lane << 4);
            unsigned l0 = ldsBase + (unsigned)(j * 1024 + (lane << 4));
            asm volatile("global_load_async_to_lds_b128 %0, %1, off"
                         :: "v"(l0), "v"(g0) : "memory");
            asm volatile("global_load_async_to_lds_b128 %0, %1, off offset:512"
                         :: "v"(l0), "v"(g0) : "memory");
        }
    }

    v8f acc0 = {}, acc1 = {}, acc2 = {}, acc3 = {};
    for (int kc = 0; kc < nkc; ++kc) {
        if (wave == 0) {
            if (kc + 1 < nkc) {
                int kn  = kc + 1;
                unsigned buf = (unsigned)(kn & 1) * 4096u;
#pragma unroll
                for (int j = 0; j < 4; ++j) {
                    uint64_t g0 = (uint64_t)(uintptr_t)(bbase +
                                  ((((size_t)(ng * 4 + j) * nkc) + kn) << 10)) +
                                  (unsigned)(lane << 4);
                    unsigned l0 = ldsBase + buf + (unsigned)(j * 1024 + (lane << 4));
                    asm volatile("global_load_async_to_lds_b128 %0, %1, off"
                                 :: "v"(l0), "v"(g0) : "memory");
                    asm volatile("global_load_async_to_lds_b128 %0, %1, off offset:512"
                                 :: "v"(l0), "v"(g0) : "memory");
                }
                asm volatile("s_wait_asynccnt 0x8" ::: "memory");  // chunk kc landed
            } else {
                asm volatile("s_wait_asynccnt 0x0" ::: "memory");
            }
        }
        __syncthreads();   // buffer kc&1 ready for all 8 waves

        v16bf a = load_a_frag(arow, kc);
        if (kc + 1 < nkc) __builtin_prefetch(arow + (kc + 1) * 32, 0, 1);
        const char* bbuf = ldsB + (kc & 1) * 4096 + (lane << 5);
        v16bf b0 = *(const v16bf*)(bbuf);
        v16bf b1 = *(const v16bf*)(bbuf + 1024);
        v16bf b2 = *(const v16bf*)(bbuf + 2048);
        v16bf b3 = *(const v16bf*)(bbuf + 3072);
        acc0 = __builtin_amdgcn_wmma_f32_16x16x32_bf16(false, a, false, b0, (short)0, acc0, false, false);
        acc1 = __builtin_amdgcn_wmma_f32_16x16x32_bf16(false, a, false, b1, (short)0, acc1, false, false);
        acc2 = __builtin_amdgcn_wmma_f32_16x16x32_bf16(false, a, false, b2, (short)0, acc2, false, false);
        acc3 = __builtin_amdgcn_wmma_f32_16x16x32_bf16(false, a, false, b3, (short)0, acc3, false, false);

        __syncthreads();   // all reads of buffer kc&1 done before reuse
    }

    int rbase = mt * 16 + ((lane >> 4) << 3);   // C VGPR v -> row rbase+v
#pragma unroll
    for (int j = 0; j < 4; ++j) {
        int col  = (ng * 4 + j) * 16 + (lane & 15);
        float bv = bias ? bias[col] : 0.0f;
        v8f acc  = (j == 0) ? acc0 : (j == 1) ? acc1 : (j == 2) ? acc2 : acc3;
#pragma unroll
        for (int v = 0; v < 8; ++v)
            C[(size_t)(rbase + v) * ldc + col] = acc[v] + bv;
    }
}

// A-fragment (M=16 padded, rows 8..15 zero) built from f32 h in LDS.
__device__ __forceinline__ v16bf build_a_lds(const float* __restrict__ buf, int kc, int lane) {
    v16bf a;
    int m = lane & 15;
    if (m < BATCH) {
        const float* p = buf + m * N_HID + kc * 32 + ((lane >> 4) << 3);
#pragma unroll
        for (int i = 0; i < 8; ++i) { a[i] = (bf16_t)p[i]; a[8 + i] = (bf16_t)p[16 + i]; }
    } else {
#pragma unroll
        for (int i = 0; i < 16; ++i) a[i] = (bf16_t)0.0f;
    }
    return a;
}

// ---------------------------------------------------------------------------
// Sequential GRU scan.  One workgroup, 32 waves; h[8][1024] f32 in LDS.
// Ax holds precomputed x_t-side gate activations, columns [z | r | hbar].
// WhP holds h-side gate weights packed, tile regions [z:0-63 | r:64-127 | h:128-191].
// Phase 1: r = sigmoid(h@WrH + AxR + br); rh = r*h -> LDS.
// Phase 2: per N-tile, z and hbar WMMA together; h' = (1-z)h + z*tanh(...)
// ---------------------------------------------------------------------------
__global__ void __launch_bounds__(1024, 1)
gru_scan_kernel(const float* __restrict__ Ax, const bf16_t* __restrict__ WhP,
                const float* __restrict__ start,
                const float* __restrict__ bz, const float* __restrict__ br,
                const float* __restrict__ bh, bf16_t* __restrict__ hidB) {
    __shared__ float hsm[BATCH * N_HID];
    __shared__ float rhsm[BATCH * N_HID];
    int tid = threadIdx.x;
    for (int o = tid; o < BATCH * N_HID; o += 1024) hsm[o] = start[o & (N_HID - 1)];
    __syncthreads();

    int lane = tid & 31;
    int wave = tid >> 5;
    const int nkc = N_HID / 32;  // 32

    for (int t = 0; t < SEQ; ++t) {
        // ---- phase 1: r gate, store r*h ----
        for (int tt = 0; tt < 2; ++tt) {
            int nt = wave * 2 + tt;                       // 0..63
            const bf16_t* bptr = WhP + (((size_t)(64 + nt) * nkc) << 9) + (lane << 4);
            v8f acc = {};
            for (int kc = 0; kc < nkc; ++kc) {
                v16bf a = build_a_lds(hsm, kc, lane);
                v16bf b = *(const v16bf*)(bptr + ((size_t)kc << 9));
                acc = __builtin_amdgcn_wmma_f32_16x16x32_bf16(false, a, false, b,
                                                              (short)0, acc, false, false);
            }
            int col   = nt * 16 + (lane & 15);
            int mbase = (lane >> 4) << 3;
#pragma unroll
            for (int v = 0; v < 8; ++v) {
                int m = mbase + v;
                if (m < BATCH) {
                    float g = acc[v] + Ax[(size_t)(m * SEQ + t) * G3 + N_HID + col] + br[col];
                    float r = 1.0f / (1.0f + __expf(-g));
                    rhsm[m * N_HID + col] = r * hsm[m * N_HID + col];
                }
            }
        }
        __syncthreads();

        // ---- phase 2: z and hbar, new h into registers ----
        float hnew[2][8];
        for (int tt = 0; tt < 2; ++tt) {
            int nt = wave * 2 + tt;
            const bf16_t* bzp = WhP + (((size_t)nt * nkc) << 9) + (lane << 4);
            const bf16_t* bhp = WhP + (((size_t)(128 + nt) * nkc) << 9) + (lane << 4);
            v8f accz = {}; v8f acch = {};
            for (int kc = 0; kc < nkc; ++kc) {
                v16bf ah = build_a_lds(hsm, kc, lane);
                v16bf ar = build_a_lds(rhsm, kc, lane);
                v16bf b0 = *(const v16bf*)(bzp + ((size_t)kc << 9));
                v16bf b1 = *(const v16bf*)(bhp + ((size_t)kc << 9));
                accz = __builtin_amdgcn_wmma_f32_16x16x32_bf16(false, ah, false, b0,
                                                               (short)0, accz, false, false);
                acch = __builtin_amdgcn_wmma_f32_16x16x32_bf16(false, ar, false, b1,
                                                               (short)0, acch, false, false);
            }
            int col   = nt * 16 + (lane & 15);
            int mbase = (lane >> 4) << 3;
#pragma unroll
            for (int v = 0; v < 8; ++v) {
                int m = mbase + v;
                if (m < BATCH) {
                    float gz = accz[v] + Ax[(size_t)(m * SEQ + t) * G3 + col] + bz[col];
                    float z  = 1.0f / (1.0f + __expf(-gz));
                    float gh = acch[v] + Ax[(size_t)(m * SEQ + t) * G3 + 2 * N_HID + col] + bh[col];
                    float hb = tanhf(gh);
                    hnew[tt][v] = (1.0f - z) * hsm[m * N_HID + col] + z * hb;
                } else hnew[tt][v] = 0.0f;
            }
        }
        __syncthreads();   // all reads of hsm/rhsm complete

        for (int tt = 0; tt < 2; ++tt) {
            int nt = wave * 2 + tt;
            int col   = nt * 16 + (lane & 15);
            int mbase = (lane >> 4) << 3;
#pragma unroll
            for (int v = 0; v < 8; ++v) {
                int m = mbase + v;
                if (m < BATCH) {
                    hsm[m * N_HID + col] = hnew[tt][v];
                    hidB[(size_t)(m * SEQ + t) * N_HID + col] = (bf16_t)hnew[tt][v];
                }
            }
        }
        __syncthreads();
    }
}

// ---------------------------------------------------------------------------
// Cross-entropy: one block per row, streaming max then sum-exp.
// ---------------------------------------------------------------------------
__global__ void loss_init_kernel(float* loss) { loss[0] = 0.0f; }

__global__ void loss_kernel(const float* __restrict__ logits, const int* __restrict__ targets,
                            float* __restrict__ loss) {
    int row = blockIdx.x;
    const float* lp = logits + (size_t)row * VOCAB;
    __shared__ float red[256];
    int tid = threadIdx.x;

    float mx = -INFINITY;
    for (int i = tid; i < VOCAB; i += 256) mx = fmaxf(mx, lp[i]);
    red[tid] = mx; __syncthreads();
    for (int s = 128; s > 0; s >>= 1) {
        if (tid < s) red[tid] = fmaxf(red[tid], red[tid + s]);
        __syncthreads();
    }
    mx = red[0]; __syncthreads();

    float sum = 0.0f;
    for (int i = tid; i < VOCAB; i += 256) sum += __expf(lp[i] - mx);
    red[tid] = sum; __syncthreads();
    for (int s = 128; s > 0; s >>= 1) {
        if (tid < s) red[tid] += red[tid + s];
        __syncthreads();
    }
    if (tid == 0) {
        int tg = targets[row];
        float l = -(lp[tg] - mx - __logf(red[0]));
        atomicAdd(loss, l * (1.0f / (float)BT));
    }
}

// ---------------------------------------------------------------------------
extern "C" void kernel_launch(void* const* d_in, const int* in_sizes, int n_in,
                              void* d_out, int out_size, void* d_ws, size_t ws_size,
                              hipStream_t stream) {
    (void)in_sizes; (void)n_in; (void)out_size; (void)ws_size;
    const int*   idx     = (const int*)d_in[0];
    const int*   targets = (const int*)d_in[1];
    const float* wte     = (const float*)d_in[2];
    const float* start   = (const float*)d_in[3];
    const float* Wz      = (const float*)d_in[4];
    const float* bz      = (const float*)d_in[5];
    const float* Wr      = (const float*)d_in[6];
    const float* br      = (const float*)d_in[7];
    const float* Wh      = (const float*)d_in[8];
    const float* bh      = (const float*)d_in[9];
    const float* lmW     = (const float*)d_in[10];
    const float* lmb     = (const float*)d_in[11];

    float* out_logits = (float*)d_out;
    float* out_loss   = out_logits + (size_t)BT * VOCAB;

    char* ws = (char*)d_ws;
    size_t off = 0;
    auto salloc = [&](size_t bytes) -> char* {
        char* p = ws + off;
        off = (off + bytes + 255) & ~(size_t)255;
        return p;
    };
    bf16_t* embB = (bf16_t*)salloc((size_t)BT * N_EMBD * 2);          //  4 MB
    bf16_t* WxP  = (bf16_t*)salloc((size_t)3 * N_EMBD * N_HID * 2);   //  3 MB
    bf16_t* WhP  = (bf16_t*)salloc((size_t)3 * N_HID * N_HID * 2);    //  6 MB
    bf16_t* lmWP = (bf16_t*)salloc((size_t)N_HID * VOCAB * 2);        // 65.5 MB
    float*  Ax   = (float*) salloc((size_t)BT * G3 * 4);              // 50 MB
    bf16_t* hidB = (bf16_t*)salloc((size_t)BT * N_HID * 2);           //  8 MB

    const int thr = 256;
    // Pack x-side gate weights (rows 0..511 of Wz/Wr/Wh) -> WxP regions [z|r|h]
    {
        size_t n1 = (size_t)N_EMBD * N_HID;
        pack_b_kernel<<<(n1 + thr - 1) / thr, thr, 0, stream>>>(Wz, N_HID, N_EMBD, N_HID, WxP);
        pack_b_kernel<<<(n1 + thr - 1) / thr, thr, 0, stream>>>(Wr, N_HID, N_EMBD, N_HID, WxP + n1);
        pack_b_kernel<<<(n1 + thr - 1) / thr, thr, 0, stream>>>(Wh, N_HID, N_EMBD, N_HID, WxP + 2 * n1);
    }
    // Pack h-side gate weights (rows 512..1535) -> WhP regions [z|r|h]
    {
        size_t n2 = (size_t)N_HID * N_HID;
        const size_t hoff = (size_t)N_EMBD * N_HID;
        pack_b_kernel<<<(n2 + thr - 1) / thr, thr, 0, stream>>>(Wz + hoff, N_HID, N_HID, N_HID, WhP);
        pack_b_kernel<<<(n2 + thr - 1) / thr, thr, 0, stream>>>(Wr + hoff, N_HID, N_HID, N_HID, WhP + n2);
        pack_b_kernel<<<(n2 + thr - 1) / thr, thr, 0, stream>>>(Wh + hoff, N_HID, N_HID, N_HID, WhP + 2 * n2);
    }
    // Pack LM head
    {
        size_t n3 = (size_t)N_HID * VOCAB;
        pack_b_kernel<<<(n3 + thr - 1) / thr, thr, 0, stream>>>(lmW, VOCAB, N_HID, VOCAB, lmWP);
    }
    // Embedding gather
    embed_kernel<<<((size_t)BT * N_EMBD) / thr, thr, 0, stream>>>(idx, wte, embB);

    // Ax = embB @ [WzX|WrX|WhX] : M=4096, N=3072, K=512
    // Mtiles=256 (32 groups of 8), Ngroups=48, nkc=16 -> 1536 blocks exact
    gemm_wmma_staged_kernel<<<(256 / 8) * (G3 / 64), 256, 0, stream>>>(
        embB, N_EMBD, WxP, nullptr, Ax, G3, BT / 16, G3 / 64, N_EMBD / 32);

    // Sequential GRU scan (single workgroup, 32 waves)
    gru_scan_kernel<<<1, 1024, 0, stream>>>(Ax, WhP, start, bz, br, bh, hidB);

    // Logits = hidB @ lmW + lmb : M=4096, N=32000, K=1024
    // Mtiles=256 (32 groups of 8), Ngroups=500, nkc=32 -> 16000 blocks exact
    gemm_wmma_staged_kernel<<<(256 / 8) * (VOCAB / 64), 256, 0, stream>>>(
        hidB, N_HID, lmWP, lmb, out_logits, VOCAB, BT / 16, VOCAB / 64, N_HID / 32);

    // Cross-entropy loss
    loss_init_kernel<<<1, 1, 0, stream>>>(out_loss);
    loss_kernel<<<BT, 256, 0, stream>>>(out_logits, targets, out_loss);
}
